// Classifier_e_GNN_large_23682449670466
// MI455X (gfx1250) — matrix operand order repaired and checked by reading, not compiled, not run on hardware
//
#include <hip/hip_runtime.h>
#include <math.h>

// ---------------------------------------------------------------------------
// CDNA5 (gfx1250) wave32 WMMA types
// ---------------------------------------------------------------------------
typedef _Float16 v16h __attribute__((ext_vector_type(16)));
typedef _Float16 v8h  __attribute__((ext_vector_type(8)));
typedef float    v8f  __attribute__((ext_vector_type(8)));

#define TM 64
#define TN 64
#define TK 32
#define SA 40   // padded LDS row stride (halves) for A  (80B: 16B-aligned, bank-spread)
#define SB 40   // padded LDS col stride (halves) for B^T

__device__ __forceinline__ v8f wmma_f16(v16h a, v16h b, v8f c) {
  return __builtin_amdgcn_wmma_f32_16x16x32_f16(false, a, false, b, (short)0, c,
                                                false, false);
}

__device__ __forceinline__ float sigmoidf_(float x) {
  return 1.f / (1.f + __expf(-x));
}

__device__ __forceinline__ void atomicMaxF(float* addr, float val) {
  if (val >= 0.f) atomicMax((int*)addr, __float_as_int(val));
  else            atomicMin((unsigned int*)addr, __float_as_uint(val));
}

// Build a 16-half fragment from two aligned 16-byte LDS chunks.
__device__ __forceinline__ v16h frag_from(const _Float16* p0, const _Float16* p1) {
  v8h a = *(const v8h*)p0;
  v8h b = *(const v8h*)p1;
  return __builtin_shufflevector(a, b, 0, 1, 2, 3, 4, 5, 6, 7,
                                 8, 9, 10, 11, 12, 13, 14, 15);
}

// ---------------------------------------------------------------------------
// Tiled WMMA GEMM: C[M,N] = A_f16[M,K](lda) * B_f16[K,N](ldb) (+bias)(+act)
// 128 threads = 4 waves (2x2); each wave owns a 32x32 tile -> 4 WMMAs/K-step.
// Requires lda%8==0 and ldb%8==0 for the vectorized fast path (all callers
// guarantee this; the 513-col graph matrix is padded to stride 520).
// act: 0=none, 1=relu, 2=sigmoid
// ---------------------------------------------------------------------------
__global__ __launch_bounds__(128)
void wmma_gemm(const _Float16* __restrict__ A, int lda,
               const _Float16* __restrict__ Bm, int ldb,
               float* __restrict__ C, int ldc,
               int M, int N, int K,
               const float* __restrict__ bias, int act) {
  __shared__ _Float16 As[TM * SA];
  __shared__ _Float16 BsT[TN * SB];   // BsT[c*SB + r] = B[k0+r][bn+c]
  const int bm = blockIdx.x * TM, bn = blockIdx.y * TN;
  const int tid = threadIdx.x;
  const int lane = tid & 31, wv = tid >> 5;
  const int mr = (wv >> 1) * 32, nc = (wv & 1) * 32;
  const int l = lane & 15, hi = lane >> 4;
  const bool mfull = (bm + TM <= M);
  const bool nfull = (bn + TN <= N);
  v8f acc00 = {}, acc01 = {}, acc10 = {}, acc11 = {};
  for (int k0 = 0; k0 < K; k0 += TK) {
    const bool kfull = (k0 + TK <= K);
    // ---- stage A tile: 64x32 halves, 16 per thread -----------------------
#pragma unroll
    for (int it = 0; it < 2; ++it) {
      int u = tid + it * 128;
      int row = u >> 2, cc8 = (u & 3) * 8;
      if (mfull && kfull) {
        v8h v = *(const v8h*)(A + (size_t)(bm + row) * lda + k0 + cc8);
        *(v8h*)&As[row * SA + cc8] = v;
      } else {
        v8h v;
        int gr = bm + row;
#pragma unroll
        for (int j = 0; j < 8; ++j) {
          int gc = k0 + cc8 + j;
          v[j] = (gr < M && gc < K) ? A[(size_t)gr * lda + gc] : (_Float16)0.f;
        }
        *(v8h*)&As[row * SA + cc8] = v;
      }
    }
    // ---- stage B tile transposed: 32x64 -> BsT ---------------------------
#pragma unroll
    for (int it = 0; it < 2; ++it) {
      int u = tid + it * 128;
      int r = u >> 3, cc8 = (u & 7) * 8;
      if (nfull && kfull) {
        v8h v = *(const v8h*)(Bm + (size_t)(k0 + r) * ldb + bn + cc8);
#pragma unroll
        for (int j = 0; j < 8; ++j) BsT[(cc8 + j) * SB + r] = v[j];
      } else {
        int gr = k0 + r;
#pragma unroll
        for (int j = 0; j < 8; ++j) {
          int gc = bn + cc8 + j;
          BsT[(cc8 + j) * SB + r] =
              (gr < K && gc < N) ? Bm[(size_t)gr * ldb + gc] : (_Float16)0.f;
        }
      }
    }
    __syncthreads();
    if (kfull && mfull && (k0 + TK < K))
      __builtin_prefetch(&A[(size_t)(bm + (tid & (TM - 1))) * lda + k0 + TK], 0, 1);
    // ---- fragments: two aligned ds_load_b128 each ------------------------
    const _Float16* a0 = &As[(mr + l) * SA];
    const _Float16* a1 = &As[(mr + 16 + l) * SA];
    v16h af0 = frag_from(a0 + hi * 16, a0 + hi * 16 + 8);
    v16h af1 = frag_from(a1 + hi * 16, a1 + hi * 16 + 8);
    const _Float16* b0 = &BsT[(nc + l) * SB + hi * 16];
    const _Float16* b1 = &BsT[(nc + 16 + l) * SB + hi * 16];
    v16h bf0 = frag_from(b0, b0 + 8);
    v16h bf1 = frag_from(b1, b1 + 8);
    acc00 = wmma_f16(af0, bf0, acc00);
    acc01 = wmma_f16(af0, bf1, acc01);
    acc10 = wmma_f16(af1, bf0, acc10);
    acc11 = wmma_f16(af1, bf1, acc11);
    __syncthreads();
  }
  // ---- epilogue ----------------------------------------------------------
#pragma unroll
  for (int mi = 0; mi < 2; ++mi) {
#pragma unroll
    for (int ni = 0; ni < 2; ++ni) {
      v8f acc = (mi == 0) ? (ni == 0 ? acc00 : acc01) : (ni == 0 ? acc10 : acc11);
      int col = bn + nc + ni * 16 + l;
      if (col >= N) continue;
      float bv = bias ? bias[col] : 0.f;
#pragma unroll
      for (int r = 0; r < 8; ++r) {
        int row = bm + mr + mi * 16 + hi * 8 + r;
        if (row >= M) continue;
        float v = acc[r] + bv;
        if (act == 1) v = fmaxf(v, 0.f);
        else if (act == 2) v = sigmoidf_(v);
        C[(size_t)row * ldc + col] = v;
      }
    }
  }
}

// ---------------------------------------------------------------------------
// Implicit-im2col WMMA conv3d (k=3, pad=1, stride=1), same tiling as GEMM.
// in: chlast? [BT,D,H,W,Cin] : [BT,Cin,D,H,W]. out: [P,O] channel-last.
// Wt f16 [K=Cin*27, O], K index = ((ci*3+kd)*3+kh)*3+kw.
// ---------------------------------------------------------------------------
__global__ __launch_bounds__(128)
void wmma_conv(const float* __restrict__ in, int chlast,
               int BT, int Cin, int D, int H, int W,
               const _Float16* __restrict__ Wt,
               const float* __restrict__ bias,
               float* __restrict__ out, int O) {
  const int P = BT * D * H * W;
  const int K = Cin * 27;
  __shared__ _Float16 As[TM * SA];
  __shared__ _Float16 BsT[TN * SB];
  const int bm = blockIdx.x * TM, bn = blockIdx.y * TN;
  const int tid = threadIdx.x;
  const int lane = tid & 31, wv = tid >> 5;
  const int mr = (wv >> 1) * 32, nc = (wv & 1) * 32;
  const int l = lane & 15, hi = lane >> 4;
  const bool nfull = (bn + TN <= O);
  v8f acc00 = {}, acc01 = {}, acc10 = {}, acc11 = {};
  for (int k0 = 0; k0 < K; k0 += TK) {
    const bool kfull = (k0 + TK <= K);
    // ---- gather A tile (implicit im2col) ---------------------------------
#pragma unroll
    for (int it = 0; it < 2; ++it) {
      int u = tid + it * 128;
      int row = u >> 2, cc8 = (u & 3) * 8;
      int p = bm + row;
      v8h vv;
      int ow = 0, oh = 0, od = 0, bt = 0;
      if (p < P) {
        ow = p % W; int t1 = p / W;
        oh = t1 % H; int t2 = t1 / H;
        od = t2 % D; bt = t2 / D;
      }
#pragma unroll
      for (int j = 0; j < 8; ++j) {
        int k = k0 + cc8 + j;
        float v = 0.f;
        if (p < P && k < K) {
          int kw = k % 3;  int t3 = k / 3;
          int kh = t3 % 3; int t4 = t3 / 3;
          int kd = t4 % 3; int ci = t4 / 3;
          int id = od + kd - 1, ih = oh + kh - 1, iw = ow + kw - 1;
          if (id >= 0 && id < D && ih >= 0 && ih < H && iw >= 0 && iw < W) {
            size_t idx = chlast
              ? ((((size_t)bt * D + id) * H + ih) * W + iw) * Cin + ci
              : ((((size_t)bt * Cin + ci) * D + id) * H + ih) * W + iw;
            v = in[idx];
          }
        }
        vv[j] = (_Float16)v;
      }
      *(v8h*)&As[row * SA + cc8] = vv;
    }
    // ---- stage weights transposed ----------------------------------------
#pragma unroll
    for (int it = 0; it < 2; ++it) {
      int u = tid + it * 128;
      int r = u >> 3, cc8 = (u & 7) * 8;
      if (nfull && kfull) {
        v8h v = *(const v8h*)(Wt + (size_t)(k0 + r) * O + bn + cc8);
#pragma unroll
        for (int j = 0; j < 8; ++j) BsT[(cc8 + j) * SB + r] = v[j];
      } else {
        int gr = k0 + r;
#pragma unroll
        for (int j = 0; j < 8; ++j) {
          int gc = bn + cc8 + j;
          BsT[(cc8 + j) * SB + r] =
              (gr < K && gc < O) ? Wt[(size_t)gr * O + gc] : (_Float16)0.f;
        }
      }
    }
    __syncthreads();
    const _Float16* a0 = &As[(mr + l) * SA];
    const _Float16* a1 = &As[(mr + 16 + l) * SA];
    v16h af0 = frag_from(a0 + hi * 16, a0 + hi * 16 + 8);
    v16h af1 = frag_from(a1 + hi * 16, a1 + hi * 16 + 8);
    const _Float16* b0 = &BsT[(nc + l) * SB + hi * 16];
    const _Float16* b1 = &BsT[(nc + 16 + l) * SB + hi * 16];
    v16h bf0 = frag_from(b0, b0 + 8);
    v16h bf1 = frag_from(b1, b1 + 8);
    acc00 = wmma_f16(af0, bf0, acc00);
    acc01 = wmma_f16(af0, bf1, acc01);
    acc10 = wmma_f16(af1, bf0, acc10);
    acc11 = wmma_f16(af1, bf1, acc11);
    __syncthreads();
  }
#pragma unroll
  for (int mi = 0; mi < 2; ++mi) {
#pragma unroll
    for (int ni = 0; ni < 2; ++ni) {
      v8f acc = (mi == 0) ? (ni == 0 ? acc00 : acc01) : (ni == 0 ? acc10 : acc11);
      int col = bn + nc + ni * 16 + l;
      if (col >= O) continue;
      float bv = bias[col];
#pragma unroll
      for (int r = 0; r < 8; ++r) {
        int row = bm + mr + mi * 16 + hi * 8 + r;
        if (row >= P) continue;
        out[(size_t)row * O + col] = acc[r] + bv;
      }
    }
  }
}

// ---------------------------------------------------------------------------
// Support kernels
// ---------------------------------------------------------------------------
__global__ void k_transpose_f16(const float* __restrict__ src, int R, int C,
                                _Float16* __restrict__ dst) {
  size_t i = (size_t)blockIdx.x * blockDim.x + threadIdx.x;
  if (i >= (size_t)R * C) return;
  int r = (int)(i / C), c = (int)(i % C);
  dst[(size_t)c * R + r] = (_Float16)src[i];
}

__global__ void k_cvt_f16(const float* __restrict__ src, size_t n,
                          _Float16* __restrict__ dst) {
  size_t i = (size_t)blockIdx.x * blockDim.x + threadIdx.x;
  if (i < n) dst[i] = (_Float16)src[i];
}

__global__ void k_fill(float* p, size_t n, float v) {
  size_t i = (size_t)blockIdx.x * blockDim.x + threadIdx.x;
  if (i < n) p[i] = v;
}

// Per-column sum / sumsq over R rows (C <= 2048) via LDS atomics.
__global__ __launch_bounds__(256)
void k_colstats(const float* __restrict__ x, size_t R, int C,
                float* __restrict__ sums, float* __restrict__ sqs) {
  __shared__ float ls[2048];
  __shared__ float lq[2048];
  for (int i = threadIdx.x; i < C; i += blockDim.x) { ls[i] = 0.f; lq[i] = 0.f; }
  __syncthreads();
  size_t total = R * (size_t)C;
  for (size_t i = (size_t)blockIdx.x * blockDim.x + threadIdx.x; i < total;
       i += (size_t)gridDim.x * blockDim.x) {
    float v = x[i];
    int c = (int)(i % C);
    atomicAdd(&ls[c], v);
    atomicAdd(&lq[c], v * v);
  }
  __syncthreads();
  for (int i = threadIdx.x; i < C; i += blockDim.x) {
    atomicAdd(&sums[i], ls[i]);
    atomicAdd(&sqs[i], lq[i]);
  }
}

// outh (optional) is written with row stride ohStride (>= C) to allow padding.
__global__ void k_bn_apply(const float* __restrict__ x, size_t R, int C,
                           const float* __restrict__ sums, const float* __restrict__ sqs,
                           const float* __restrict__ g, const float* __restrict__ b,
                           int relu, float* __restrict__ outf,
                           _Float16* __restrict__ outh, int ohStride) {
  size_t i = (size_t)blockIdx.x * blockDim.x + threadIdx.x;
  if (i >= R * (size_t)C) return;
  int c = (int)(i % C);
  size_t r = i / C;
  float invR = 1.f / (float)R;
  float m = sums[c] * invR;
  float v = sqs[c] * invR - m * m;
  float y = (x[i] - m) * rsqrtf(v + 1e-5f) * g[c] + b[c];
  if (relu) y = fmaxf(y, 0.f);
  if (outf) outf[i] = y;
  if (outh) outh[r * (size_t)ohStride + c] = (_Float16)y;
}

// channel-last 2x2x2 stride-2 max pool with padding; cmajor picks NCDHW flatten
__global__ void k_maxpool(const float* __restrict__ in, int BT, int C,
                          int D, int H, int W, int pd, int ph, int pw,
                          int oD, int oH, int oW, int cmajor,
                          float* __restrict__ out) {
  size_t total = (size_t)BT * oD * oH * oW * C;
  size_t i = (size_t)blockIdx.x * blockDim.x + threadIdx.x;
  if (i >= total) return;
  int c = (int)(i % C);
  size_t r = i / C;
  int ow = (int)(r % oW); r /= oW;
  int oh = (int)(r % oH); r /= oH;
  int od = (int)(r % oD);
  int bt = (int)(r / oD);
  float m = -INFINITY;
  for (int kd = 0; kd < 2; ++kd) {
    int id = od * 2 - pd + kd; if (id < 0 || id >= D) continue;
    for (int kh = 0; kh < 2; ++kh) {
      int ih = oh * 2 - ph + kh; if (ih < 0 || ih >= H) continue;
      for (int kw = 0; kw < 2; ++kw) {
        int iw = ow * 2 - pw + kw; if (iw < 0 || iw >= W) continue;
        float v = in[((((size_t)bt * D + id) * H + ih) * W + iw) * C + c];
        m = fmaxf(m, v);
      }
    }
  }
  size_t oi = cmajor
    ? ((size_t)bt * C + c) * ((size_t)oD * oH * oW) + ((size_t)od * oH + oh) * oW + ow
    : ((((size_t)bt * oD + od) * oH + oh) * oW + ow) * C + c;
  out[oi] = m;
}

__global__ void k_gru_gate(const float* __restrict__ gi, const float* __restrict__ gh,
                           float* __restrict__ h, _Float16* __restrict__ hh,
                           _Float16* __restrict__ ys, int Bb, int Hd, int t, int T) {
  int i = blockIdx.x * blockDim.x + threadIdx.x;
  if (i >= Bb * Hd) return;
  int b = i / Hd, c = i % Hd;
  const float* gib = gi + (size_t)b * 3 * Hd;
  const float* ghb = gh + (size_t)b * 3 * Hd;
  float r = sigmoidf_(gib[c] + ghb[c]);
  float z = sigmoidf_(gib[Hd + c] + ghb[Hd + c]);
  float n = tanhf(gib[2 * Hd + c] + r * ghb[2 * Hd + c]);
  float hn = (1.f - z) * n + z * h[i];
  h[i] = hn;
  hh[i] = (_Float16)hn;
  ys[((size_t)b * T + t) * Hd + c] = (_Float16)hn;
}

__global__ void k_build_xg(const float* __restrict__ z, const int* __restrict__ low_res,
                           const int* __restrict__ t_list, const int* __restrict__ kmat,
                           int Bk, const float* __restrict__ enc, int encC,
                           float* __restrict__ xg, int N) {
  int n = blockIdx.x * blockDim.x + threadIdx.x;
  if (n >= N) return;
  long long kn = (long long)low_res[n] * 1000000LL + (long long)t_list[n];
  int idx = -1;
  for (int b = 0; b < Bk; ++b) {
    long long kk = (long long)kmat[2 * b] * 1000000LL + (long long)kmat[2 * b + 1];
    if (kk == kn) { idx = b; break; }
  }
  float* row = xg + (size_t)n * (encC + 1);
  row[0] = z[n];
  if (idx >= 0) {
    const float* er = enc + (size_t)idx * encC;
    for (int c = 0; c < encC; ++c) row[1 + c] = er[c];
  } else {
    for (int c = 0; c < encC; ++c) row[1 + c] = 0.f;
  }
}

__global__ void k_gat_score(const float* __restrict__ xl, const float* __restrict__ xr,
                            const int* __restrict__ srcs, const int* __restrict__ dsts,
                            int E, int N, int Hh, int Cc,
                            const float* __restrict__ att,
                            float* __restrict__ esc, float* __restrict__ emax,
                            float* __restrict__ cnt) {
  size_t i = (size_t)blockIdx.x * blockDim.x + threadIdx.x;
  int Et = E + N;
  if (i >= (size_t)Et * Hh) return;
  int e = (int)(i / Hh), h = (int)(i % Hh);
  int s = e < E ? srcs[e] : (e - E);
  int d = e < E ? dsts[e] : (e - E);
  const float* xls = xl + (size_t)s * Hh * Cc + (size_t)h * Cc;
  const float* xrd = xr + (size_t)d * Hh * Cc + (size_t)h * Cc;
  const float* ah = att + (size_t)h * Cc;
  float acc = 0.f;
  for (int c = 0; c < Cc; ++c) {
    float m = xls[c] + xrd[c];
    m = m > 0.f ? m : 0.2f * m;
    acc += ah[c] * m;
  }
  esc[i] = acc;
  atomicMaxF(&emax[(size_t)d * Hh + h], acc);
  if (h == 0) atomicAdd(&cnt[d], 1.f);
}

__global__ void k_gat_exp(float* __restrict__ esc, const int* __restrict__ dsts,
                          int E, int N, int Hh, const float* __restrict__ emax,
                          float* __restrict__ den) {
  size_t i = (size_t)blockIdx.x * blockDim.x + threadIdx.x;
  int Et = E + N;
  if (i >= (size_t)Et * Hh) return;
  int e = (int)(i / Hh), h = (int)(i % Hh);
  int d = e < E ? dsts[e] : (e - E);
  float ex = __expf(esc[i] - emax[(size_t)d * Hh + h]);
  esc[i] = ex;
  atomicAdd(&den[(size_t)d * Hh + h], ex);
}

__global__ void k_gat_agg(const float* __restrict__ esc, const float* __restrict__ xl,
                          const int* __restrict__ srcs, const int* __restrict__ dsts,
                          int E, int N, int Hh, int Cc,
                          const float* __restrict__ den, float* __restrict__ msum) {
  size_t i = (size_t)blockIdx.x * blockDim.x + threadIdx.x;
  int HC = Hh * Cc;
  int Et = E + N;
  if (i >= (size_t)Et * HC) return;
  int e = (int)(i / HC), hc = (int)(i % HC), h = hc / Cc;
  int s = e < E ? srcs[e] : (e - E);
  int d = e < E ? dsts[e] : (e - E);
  float alpha = esc[(size_t)e * Hh + h] / den[(size_t)d * Hh + h];
  atomicAdd(&msum[(size_t)d * HC + hc], alpha * xl[(size_t)s * HC + hc]);
}

__global__ void k_gat_fin(const float* __restrict__ msum, const float* __restrict__ cnt,
                          const float* __restrict__ bias, float* __restrict__ out,
                          int N, int HC) {
  size_t i = (size_t)blockIdx.x * blockDim.x + threadIdx.x;
  if (i >= (size_t)N * HC) return;
  int n = (int)(i / HC), j = (int)(i % HC);
  out[i] = msum[i] / cnt[n] + bias[j];
}

__global__ void k_final(const float* __restrict__ xg3, const float* __restrict__ y,
                        const int* __restrict__ tidx, int NT, float* __restrict__ out) {
  int i = blockIdx.x * blockDim.x + threadIdx.x;
  if (i >= NT) return;
  int n = tidx[i];
  out[i] = sigmoidf_(xg3[n]);
  out[NT + i] = y[n];
}

// ---------------------------------------------------------------------------
// Host orchestration
// ---------------------------------------------------------------------------
static inline void launch_gemm(const _Float16* A, int lda, const _Float16* Bm, int ldb,
                               float* C, int ldc, int M, int Nn, int K,
                               const float* bias, int act, hipStream_t s) {
  dim3 grid((unsigned)((M + TM - 1) / TM), (unsigned)((Nn + TN - 1) / TN));
  wmma_gemm<<<grid, 128, 0, s>>>(A, lda, Bm, ldb, C, ldc, M, Nn, K, bias, act);
}

static inline void launch_transpose(const float* src, int R, int C, _Float16* dst,
                                    hipStream_t s) {
  size_t n = (size_t)R * C;
  k_transpose_f16<<<(unsigned)((n + 255) / 256), 256, 0, s>>>(src, R, C, dst);
}

extern "C" void kernel_launch(void* const* d_in, const int* in_sizes, int n_in,
                              void* d_out, int out_size, void* d_ws, size_t ws_size,
                              hipStream_t stream) {
  const float* in0     = (const float*)d_in[0];   // [B,25,5,5,6,6]
  const float* zIn     = (const float*)d_in[1];
  const float* yIn     = (const float*)d_in[2];
  const int*   low_res = (const int*)d_in[3];
  const int*   t_list  = (const int*)d_in[4];
  const int*   kmat    = (const int*)d_in[5];
  const int*   eidx    = (const int*)d_in[6];     // [2,E]
  const int*   tridx   = (const int*)d_in[7];
  enum {
    P_C1W = 8, P_C1B, P_G1, P_B1, P_C2W, P_C2B, P_G2, P_B2,
    P_C3W, P_C3B, P_G3, P_B3, P_FC1W, P_FC1B, P_G4, P_B4,
    P_FC2W, P_FC2B, P_G5, P_B5,
    P_WIH0, P_WHH0, P_BIH0, P_BHH0, P_WIH1, P_WHH1, P_BIH1, P_BHH1,
    P_DW, P_DB, P_GN0G, P_GN0B,
    P_A1WL, P_A1BL, P_A1WR, P_A1BR, P_A1ATT, P_A1BIAS, P_GN1G, P_GN1B,
    P_A2WL, P_A2BL, P_A2WR, P_A2BR, P_A2ATT, P_A2BIAS, P_GN2G, P_GN2B,
    P_A3WL, P_A3BL, P_A3WR, P_A3BR, P_A3ATT, P_A3BIAS
  };
  #define PP(i) ((const float*)d_in[(i)])

  int Bb = in_sizes[0] / 22500; if (Bb <= 0) Bb = 64;
  const int T = 25, BT = Bb * T;
  const int N  = in_sizes[1];
  const int E  = in_sizes[6] / 2;
  const int NT = in_sizes[7];
  const int Bk = in_sizes[5] / 2;
  const int P1 = BT * 5 * 6 * 6;
  const int P3 = BT * 3 * 4 * 4;
  const int XGP = 520;               // padded f16 row stride for 513-col matrix

  char* wsb = (char*)d_ws;
  size_t off = 0;
  auto alloc = [&](size_t bytes) -> void* {
    void* p = wsb + off;
    off = (off + bytes + 255) & ~(size_t)255;
    return p;
  };
  // persistent region
  float* sums = (float*)alloc(2048 * 4);
  float* sqs  = (float*)alloc(2048 * 4);
  _Float16* w1t   = (_Float16*)alloc((size_t)135 * 64 * 2);
  _Float16* w2t   = (_Float16*)alloc((size_t)1728 * 64 * 2);
  _Float16* w3t   = (_Float16*)alloc((size_t)1728 * 256 * 2);
  _Float16* fw1t  = (_Float16*)alloc((size_t)2048 * 512 * 2);
  _Float16* fw2t  = (_Float16*)alloc((size_t)512 * 256 * 2);
  _Float16* wih0t = (_Float16*)alloc((size_t)256 * 768 * 2);
  _Float16* whh0t = (_Float16*)alloc((size_t)256 * 768 * 2);
  _Float16* wih1t = (_Float16*)alloc((size_t)256 * 768 * 2);
  _Float16* whh1t = (_Float16*)alloc((size_t)256 * 768 * 2);
  _Float16* dwt   = (_Float16*)alloc((size_t)6400 * 512 * 2);
  float* h    = (float*)alloc((size_t)Bb * 256 * 4);
  _Float16* h16 = (_Float16*)alloc((size_t)Bb * 256 * 2);
  float* gi   = (float*)alloc((size_t)Bb * 768 * 4);
  float* gh   = (float*)alloc((size_t)Bb * 768 * 4);
  float* enc  = (float*)alloc((size_t)Bb * 512 * 4);
  _Float16* wlT1 = (_Float16*)alloc((size_t)513 * 256 * 2);
  _Float16* wrT1 = (_Float16*)alloc((size_t)513 * 256 * 2);
  _Float16* wlT2 = (_Float16*)alloc((size_t)256 * 128 * 2);
  _Float16* wrT2 = (_Float16*)alloc((size_t)256 * 128 * 2);
  _Float16* wlT3 = (_Float16*)alloc((size_t)128 * 2);
  _Float16* wrT3 = (_Float16*)alloc((size_t)128 * 2);
  const size_t arena = off;

  // phase A arena (conv / fc / gru)
  off = arena;
  float* act1  = (float*)alloc((size_t)P1 * 64 * 4);
  float* act2  = (float*)alloc((size_t)P1 * 64 * 4);
  float* pool1 = (float*)alloc((size_t)P3 * 64 * 4);
  float* act3  = (float*)alloc((size_t)P3 * 256 * 4);
  float* pool2 = (float*)alloc((size_t)BT * 2048 * 4);
  _Float16* fcin = (_Float16*)alloc((size_t)BT * 2048 * 2);
  float* fc1o  = (float*)alloc((size_t)BT * 512 * 4);
  _Float16* fc1h = (_Float16*)alloc((size_t)BT * 512 * 2);
  float* fc2o  = (float*)alloc((size_t)BT * 256 * 4);
  _Float16* xseq = (_Float16*)alloc((size_t)BT * 256 * 2);
  _Float16* ys1  = (_Float16*)alloc((size_t)BT * 256 * 2);
  _Float16* ys2  = (_Float16*)alloc((size_t)BT * 256 * 2);

  // phase B arena (graph) — overlaps phase A (stream-ordered reuse)
  off = arena;
  float* xg   = (float*)alloc((size_t)N * 513 * 4);
  _Float16* xgh = (_Float16*)alloc((size_t)N * XGP * 2);  // padded stride 520
  float* xl   = (float*)alloc((size_t)N * 256 * 4);
  float* xr   = (float*)alloc((size_t)N * 256 * 4);
  float* esc  = (float*)alloc((size_t)(E + N) * 2 * 4);
  float* emax = (float*)alloc((size_t)N * 2 * 4);
  float* den  = (float*)alloc((size_t)N * 2 * 4);
  float* cnt  = (float*)alloc((size_t)N * 4);
  float* msum = (float*)alloc((size_t)N * 256 * 4);
  float* gato = (float*)alloc((size_t)N * 256 * 4);
  _Float16* gath = (_Float16*)alloc((size_t)N * 256 * 2);
  float* xg3  = (float*)alloc((size_t)N * 4);

  auto bn = [&](const float* x, size_t R, int C, const float* gg, const float* bb,
                int relu, float* outf, _Float16* outh, int ohStride) {
    hipMemsetAsync(sums, 0, (size_t)C * 4, stream);
    hipMemsetAsync(sqs, 0, (size_t)C * 4, stream);
    k_colstats<<<512, 256, 0, stream>>>(x, R, C, sums, sqs);
    size_t tot = R * (size_t)C;
    k_bn_apply<<<(unsigned)((tot + 255) / 256), 256, 0, stream>>>(
        x, R, C, sums, sqs, gg, bb, relu, outf, outh, ohStride);
  };

  // ======================= phase A: CNN / FC / GRU ========================
  launch_transpose(PP(P_C1W), 64, 135, w1t, stream);
  {
    dim3 g((unsigned)((P1 + TM - 1) / TM), 1);
    wmma_conv<<<g, 128, 0, stream>>>(in0, 0, BT, 5, 5, 6, 6, w1t, PP(P_C1B), act1, 64);
  }
  bn(act1, (size_t)P1, 64, PP(P_G1), PP(P_B1), 1, act1, nullptr, 64);

  launch_transpose(PP(P_C2W), 64, 1728, w2t, stream);
  {
    dim3 g((unsigned)((P1 + TM - 1) / TM), 1);
    wmma_conv<<<g, 128, 0, stream>>>(act1, 1, BT, 64, 5, 6, 6, w2t, PP(P_C2B), act2, 64);
  }
  bn(act2, (size_t)P1, 64, PP(P_G2), PP(P_B2), 1, act2, nullptr, 64);

  {
    size_t np = (size_t)BT * 3 * 4 * 4 * 64;
    k_maxpool<<<(unsigned)((np + 255) / 256), 256, 0, stream>>>(
        act2, BT, 64, 5, 6, 6, 1, 1, 1, 3, 4, 4, 0, pool1);
  }

  launch_transpose(PP(P_C3W), 256, 1728, w3t, stream);
  {
    dim3 g((unsigned)((P3 + TM - 1) / TM), 4);
    wmma_conv<<<g, 128, 0, stream>>>(pool1, 1, BT, 64, 3, 4, 4, w3t, PP(P_C3B), act3, 256);
  }
  bn(act3, (size_t)P3, 256, PP(P_G3), PP(P_B3), 1, act3, nullptr, 256);

  {
    size_t np = (size_t)BT * 2 * 2 * 2 * 256;
    k_maxpool<<<(unsigned)((np + 255) / 256), 256, 0, stream>>>(
        act3, BT, 256, 3, 4, 4, 1, 0, 0, 2, 2, 2, 1, pool2);
    size_t nf = (size_t)BT * 2048;
    k_cvt_f16<<<(unsigned)((nf + 255) / 256), 256, 0, stream>>>(pool2, nf, fcin);
  }

  launch_transpose(PP(P_FC1W), 512, 2048, fw1t, stream);
  launch_gemm(fcin, 2048, fw1t, 512, fc1o, 512, BT, 512, 2048, PP(P_FC1B), 0, stream);
  bn(fc1o, (size_t)BT, 512, PP(P_G4), PP(P_B4), 1, fc1o, fc1h, 512);

  launch_transpose(PP(P_FC2W), 256, 512, fw2t, stream);
  launch_gemm(fc1h, 512, fw2t, 256, fc2o, 256, BT, 256, 512, PP(P_FC2B), 0, stream);
  bn(fc2o, (size_t)BT, 256, PP(P_G5), PP(P_B5), 1, fc2o, xseq, 256);

  // GRU layer 1
  launch_transpose(PP(P_WIH0), 768, 256, wih0t, stream);
  launch_transpose(PP(P_WHH0), 768, 256, whh0t, stream);
  hipMemsetAsync(h, 0, (size_t)Bb * 256 * 4, stream);
  hipMemsetAsync(h16, 0, (size_t)Bb * 256 * 2, stream);
  for (int t = 0; t < T; ++t) {
    launch_gemm(xseq + (size_t)t * 256, T * 256, wih0t, 768, gi, 768, Bb, 768, 256,
                PP(P_BIH0), 0, stream);
    launch_gemm(h16, 256, whh0t, 768, gh, 768, Bb, 768, 256, PP(P_BHH0), 0, stream);
    k_gru_gate<<<(Bb * 256 + 255) / 256, 256, 0, stream>>>(gi, gh, h, h16, ys1, Bb, 256, t, T);
  }
  // GRU layer 2
  launch_transpose(PP(P_WIH1), 768, 256, wih1t, stream);
  launch_transpose(PP(P_WHH1), 768, 256, whh1t, stream);
  hipMemsetAsync(h, 0, (size_t)Bb * 256 * 4, stream);
  hipMemsetAsync(h16, 0, (size_t)Bb * 256 * 2, stream);
  for (int t = 0; t < T; ++t) {
    launch_gemm(ys1 + (size_t)t * 256, T * 256, wih1t, 768, gi, 768, Bb, 768, 256,
                PP(P_BIH1), 0, stream);
    launch_gemm(h16, 256, whh1t, 768, gh, 768, Bb, 768, 256, PP(P_BHH1), 0, stream);
    k_gru_gate<<<(Bb * 256 + 255) / 256, 256, 0, stream>>>(gi, gh, h, h16, ys2, Bb, 256, t, T);
  }
  // encoder dense: [B,6400] x [6400,512], relu
  launch_transpose(PP(P_DW), 512, 6400, dwt, stream);
  launch_gemm(ys2, 6400, dwt, 512, enc, 512, Bb, 512, 6400, PP(P_DB), 1, stream);

  // ======================= phase B: graph / GATv2 =========================
  k_build_xg<<<(N + 255) / 256, 256, 0, stream>>>(zIn, low_res, t_list, kmat, Bk,
                                                  enc, 512, xg, N);
  bn(xg, (size_t)N, 513, PP(P_GN0G), PP(P_GN0B), 0, xg, xgh, XGP);

  const int* srcs = eidx;
  const int* dsts = eidx + E;
  auto run_gat = [&](const _Float16* xin, int inC, int inLd, int Hh, int Cc,
                     const float* wl, const float* bl, const float* wr, const float* br,
                     const float* att, const float* abias,
                     _Float16* wlT, _Float16* wrT, float* outp) {
    int HC = Hh * Cc;
    launch_transpose(wl, HC, inC, wlT, stream);
    launch_transpose(wr, HC, inC, wrT, stream);
    launch_gemm(xin, inLd, wlT, HC, xl, HC, N, HC, inC, bl, 0, stream);
    launch_gemm(xin, inLd, wrT, HC, xr, HC, N, HC, inC, br, 0, stream);
    hipMemsetAsync(cnt, 0, (size_t)N * 4, stream);
    hipMemsetAsync(den, 0, (size_t)N * Hh * 4, stream);
    hipMemsetAsync(msum, 0, (size_t)N * HC * 4, stream);
    size_t nm = (size_t)N * Hh;
    k_fill<<<(unsigned)((nm + 255) / 256), 256, 0, stream>>>(emax, nm, -INFINITY);
    size_t n1 = (size_t)(E + N) * Hh;
    k_gat_score<<<(unsigned)((n1 + 255) / 256), 256, 0, stream>>>(
        xl, xr, srcs, dsts, E, N, Hh, Cc, att, esc, emax, cnt);
    k_gat_exp<<<(unsigned)((n1 + 255) / 256), 256, 0, stream>>>(
        esc, dsts, E, N, Hh, emax, den);
    size_t n2 = (size_t)(E + N) * HC;
    k_gat_agg<<<(unsigned)((n2 + 255) / 256), 256, 0, stream>>>(
        esc, xl, srcs, dsts, E, N, Hh, Cc, den, msum);
    size_t n3 = (size_t)N * HC;
    k_gat_fin<<<(unsigned)((n3 + 255) / 256), 256, 0, stream>>>(
        msum, cnt, abias, outp, N, HC);
  };

  run_gat(xgh, 513, XGP, 2, 128, PP(P_A1WL), PP(P_A1BL), PP(P_A1WR), PP(P_A1BR),
          PP(P_A1ATT), PP(P_A1BIAS), wlT1, wrT1, gato);
  bn(gato, (size_t)N, 256, PP(P_GN1G), PP(P_GN1B), 1, gato, gath, 256);

  run_gat(gath, 256, 256, 1, 128, PP(P_A2WL), PP(P_A2BL), PP(P_A2WR), PP(P_A2BR),
          PP(P_A2ATT), PP(P_A2BIAS), wlT2, wrT2, gato);
  bn(gato, (size_t)N, 128, PP(P_GN2G), PP(P_GN2B), 1, gato, gath, 128);

  run_gat(gath, 128, 128, 1, 1, PP(P_A3WL), PP(P_A3BL), PP(P_A3WR), PP(P_A3BR),
          PP(P_A3ATT), PP(P_A3BIAS), wlT3, wrT3, xg3);

  k_final<<<(NT + 255) / 256, 256, 0, stream>>>(xg3, yIn, tridx, NT, (float*)d_out);
  #undef PP
  (void)n_in; (void)out_size; (void)ws_size;
}
